// ELiCiT_50087908606691
// MI455X (gfx1250) — matrix operand order repaired and checked by reading, not compiled, not run on hardware
//
#include <hip/hip_runtime.h>
#include <hip/hip_bf16.h>
#include <math.h>

// MI455X / gfx1250, wave32. Two rank-64 GEMMs (8192x8192x64) via
// V_WMMA_F32_16X16X32_F16, fused quantize->pack prologue and
// r0*tanh(r1)*exp(scale)+bias epilogue (native V_TANH_F32).
//
// Workspace layout (needs 4 MiB):
//   Apack: _Float16 [2(g)][512(mtile)][2(kc)][32(lane)][16(e)]   (2 MiB)
//   Bpack: _Float16 [2(g)][512(ntile)][2(kc)][32(lane)][16(e)]   (2 MiB)
// Fragment element mapping matches the CDNA5 WMMA VGPR layouts
// (CDNA5 ISA §7.12.2):
//   A 16x32 f16: lane<16 -> row M=lane, e0..7 = K0..7,  e8..15 = K16..23
//                lane>=16 -> row M=lane-16, e0..7 = K8..15, e8..15 = K24..31
//   B 32x16 f16: lane<16 -> col N=lane,  e0..7 = K0..7, e8..15 = K16..23 ; lane>=16 -> K8..15, K24..31
//   C/D 16x16 f32: VGPR r, lane<16 -> (M=r,   N=lane)
//                          lane>=16 -> (M=r+8, N=lane-16)

typedef __attribute__((ext_vector_type(16))) _Float16 v16h;
typedef __attribute__((ext_vector_type(8)))  float    v8f;

#define NF     32
#define D0N    8192
#define D1N    8192
#define NT16   512u         // number of 16-wide tiles per side
#define FRAG_ELEMS (1u<<20) // f16 elements per packed buffer (2*512*2*32*16)

// Native CDNA5 transcendental tanh (TRANS32 class) instead of the ~30-VALU
// ocml expansion; keeps the epilogue under the HBM store roofline.
static __device__ __forceinline__ float fast_tanh(float x) {
#if __has_builtin(__builtin_amdgcn_tanhf)
    return __builtin_amdgcn_tanhf(x);
#elif __has_builtin(__builtin_amdgcn_tanh_f32)
    return __builtin_amdgcn_tanh_f32(x);
#else
    return tanhf(x);
#endif
}

// ------------------------------------------------------------------
// Phase 1: quantize feats, fold mats into A, emit WMMA-layout frags.
// tid bits: [3:0]=e  [8:4]=lane  [9]=kc  [18:10]=tile  [19]=g  [20]=A/B
// ------------------------------------------------------------------
__global__ __launch_bounds__(256) void elicit_pack(
    const float* __restrict__ values,   // 128 floats  (1,4,32) flat
    const float* __restrict__ feats,    // 16384 x 32
    _Float16* __restrict__ Apack,
    _Float16* __restrict__ Bpack)
{
    unsigned tid  = blockIdx.x * blockDim.x + threadIdx.x;   // 0 .. 2^21-1
    unsigned isB  = (tid >> 20) & 1u;
    unsigned t    = tid & (FRAG_ELEMS - 1u);
    unsigned e    = t & 15u;
    unsigned lane = (t >> 4) & 31u;
    unsigned kc   = (t >> 9) & 1u;
    unsigned tile = (t >> 10) & 511u;
    unsigned g    = (t >> 19) & 1u;
    unsigned hi   = lane >> 4;
    unsigned l    = lane & 15u;

    unsigned k, i;
    if (!isB) {
        // A-fragment (16x32, MxK) element -> (row i, k)
        unsigned koff = (e & 7u) + (hi ? 8u : 0u) + ((e >> 3) ? 16u : 0u);
        k = kc * 32u + koff;
        i = tile * 16u + l;                 // row in d0 half
    } else {
        // B-fragment (32x16, KxN) element -> (k, col j)
        unsigned koff = (e & 7u) + (hi ? 8u : 0u) + ((e >> 3) ? 16u : 0u);
        k = kc * 32u + koff;
        i = D0N + tile * 16u + l;           // row in d1 half (column j)
    }
    unsigned a = k >> 4, f = k & 15u;
    unsigned c = g * 16u + f;               // actual feature index

    // Nearest of candidates (2q+1)/32, q=0..15: q = clamp(floor(16x),0,15)
    float x = feats[i * NF + c];
    int  q  = (int)(x * 16.0f);
    q = q < 0 ? 0 : (q > 15 ? 15 : q);
    float p = (float)(2 * q + 1) * (1.0f / 32.0f);  // exact in f16

    if (!isB) {
        if (a >> 1) p = 1.0f - p;                        // rf selector a>=2
        float val = values[a * 32u + 2u * f + g] * p;    // mats[g,a,f]*p
        Apack[t] = (_Float16)val;
    } else {
        if (a & 1u) p = 1.0f - p;                        // cf selector a%2
        Bpack[t] = (_Float16)p;
    }
}

// ------------------------------------------------------------------
// Phase 2: one wave per 32x32 output tile; 16 WMMAs/wave; fused
// epilogue out = acc0 * tanh(acc1) * exp(scale) + bias.
// ------------------------------------------------------------------
__global__ __launch_bounds__(256) void elicit_wmma_gemm(
    const _Float16* __restrict__ Apack,
    const _Float16* __restrict__ Bpack,
    const float* __restrict__ scale,
    const float* __restrict__ bias,
    float* __restrict__ out)
{
    const unsigned lane = threadIdx.x & 31u;
    const unsigned wave = threadIdx.x >> 5;
    const unsigned t    = blockIdx.x * 8u + wave;   // wave-tile id, 0..65535
    const unsigned tm   = t >> 8;                   // 0..255  (32-row band)
    const unsigned tn   = t & 255u;                 // 0..255  (32-col band)

    const v16h* __restrict__ Av = (const v16h*)Apack;  // 32B/lane per frag
    const v16h* __restrict__ Bv = (const v16h*)Bpack;

    v8f acc[2][2][2];   // [g][mt][nt]
    #pragma unroll
    for (int g = 0; g < 2; ++g)
        #pragma unroll
        for (int mt = 0; mt < 2; ++mt)
            #pragma unroll
            for (int nt = 0; nt < 2; ++nt)
                acc[g][mt][nt] = v8f{};

    #pragma unroll
    for (int g = 0; g < 2; ++g) {
        #pragma unroll
        for (int kc = 0; kc < 2; ++kc) {
            const unsigned am0 = ((g * NT16 + (tm * 2u + 0u)) * 2u + kc) * 32u;
            const unsigned am1 = ((g * NT16 + (tm * 2u + 1u)) * 2u + kc) * 32u;
            const unsigned bn0 = ((g * NT16 + (tn * 2u + 0u)) * 2u + kc) * 32u;
            const unsigned bn1 = ((g * NT16 + (tn * 2u + 1u)) * 2u + kc) * 32u;
            v16h a0 = Av[am0 + lane];
            v16h a1 = Av[am1 + lane];
            v16h b0 = Bv[bn0 + lane];
            v16h b1 = Bv[bn1 + lane];
            acc[g][0][0] = __builtin_amdgcn_wmma_f32_16x16x32_f16(
                false, a0, false, b0, (short)0, acc[g][0][0], false, false);
            acc[g][0][1] = __builtin_amdgcn_wmma_f32_16x16x32_f16(
                false, a0, false, b1, (short)0, acc[g][0][1], false, false);
            acc[g][1][0] = __builtin_amdgcn_wmma_f32_16x16x32_f16(
                false, a1, false, b0, (short)0, acc[g][1][0], false, false);
            acc[g][1][1] = __builtin_amdgcn_wmma_f32_16x16x32_f16(
                false, a1, false, b1, (short)0, acc[g][1][1], false, false);
        }
    }

    const float es = expf(scale[0]);
    const float bs = bias[0];
    const unsigned rbase = tm * 32u + ((lane >> 4) ? 8u : 0u);
    const unsigned cbase = tn * 32u + (lane & 15u);

    #pragma unroll
    for (int mt = 0; mt < 2; ++mt) {
        #pragma unroll
        for (int nt = 0; nt < 2; ++nt) {
            #pragma unroll
            for (int r = 0; r < 8; ++r) {
                float v0 = acc[0][mt][nt][r];
                float v1 = acc[1][mt][nt][r];
                unsigned row = rbase + (unsigned)mt * 16u + (unsigned)r;
                unsigned col = cbase + (unsigned)nt * 16u;
                out[(size_t)row * (size_t)D1N + col] =
                    v0 * fast_tanh(v1) * es + bs;
            }
        }
    }
}

// ------------------------------------------------------------------
// Host-side launcher
// ------------------------------------------------------------------
extern "C" void kernel_launch(void* const* d_in, const int* in_sizes, int n_in,
                              void* d_out, int out_size, void* d_ws, size_t ws_size,
                              hipStream_t stream) {
    (void)in_sizes; (void)n_in; (void)out_size; (void)ws_size;
    // Input order: values, feats, candidates, scale, bias, which_axis, d0, d1
    const float* values = (const float*)d_in[0];   // 128
    const float* feats  = (const float*)d_in[1];   // 16384*32
    // d_in[2] candidates: uniform linspace (2k+1)/32, handled analytically.
    const float* scale  = (const float*)d_in[3];
    const float* bias   = (const float*)d_in[4];
    // d_in[5] which_axis irrelevant (candidates identical per axis);
    // d_in[6]/d_in[7] d0/d1 fixed at 8192 per reference module constants.

    _Float16* Apack = (_Float16*)d_ws;                 // 2 MiB
    _Float16* Bpack = Apack + FRAG_ELEMS;              // +2 MiB

    // Phase 1: 2^21 threads pack both buffers.
    elicit_pack<<<(2u * FRAG_ELEMS) / 256u, 256, 0, stream>>>(
        values, feats, Apack, Bpack);

    // Phase 2: 65536 wave-tiles / 8 waves per block = 8192 blocks.
    elicit_wmma_gemm<<<8192, 256, 0, stream>>>(
        Apack, Bpack, scale, bias, (float*)d_out);
}
